// SwinTransformerObjectDetection_61203283968865
// MI455X (gfx1250) — compile-verified, matrix operand order checked by reading, not run on hardware
//
#include <hip/hip_runtime.h>
#include <hip/hip_bf16.h>
#include <math.h>

// ---------------------------------------------------------------------------
// CDNA5 / gfx1250 Swin-Transformer forward.
// All GEMMs run on v_wmma_f32_16x16x32_bf16 (bf16 operands, f32 accumulate).
// Wave32; 128-thread blocks = 4 waves; 64x64 output tiles per block.
// LDS tiles are stored in *fragment-major* layout so each WMMA operand is a
// single v16bf (2x ds_load_b128) per lane instead of 16 scalar ds_load_u16.
// ---------------------------------------------------------------------------

typedef __bf16 bf16_t;
typedef __attribute__((ext_vector_type(16))) __bf16 v16bf;
typedef __attribute__((ext_vector_type(8)))  float  v8f;

// ISA 7.12.2 16-bit operand K-mapping, inverted:
//   lane-half: part(k) = (k>>3)&1  (adds 16 to lane)
//   element  : e(k)    = k - 8*(k>>4) - 8*part(k)
__device__ __forceinline__ int frag_lane_add(int kk) { return ((kk >> 3) & 1) << 4; }
__device__ __forceinline__ int frag_e(int kk) {
  return kk - (((kk >> 4) & 1) << 3) - (((kk >> 3) & 1) << 3);
}

// A-operand fragment-major offset for element (r in [0,64), k in [0,32)).
__device__ __forceinline__ int a_off(int r, int kk) {
  return (r >> 4) * 512 + ((r & 15) + frag_lane_add(kk)) * 16 + frag_e(kk);
}
// B-operand fragment-major offset for element (k in [0,32), n in [0,64)).
__device__ __forceinline__ int b_off(int kk, int n) {
  return (n >> 4) * 512 + ((n & 15) + frag_lane_add(kk)) * 16 + frag_e(kk);
}

__device__ __forceinline__ float gelu_erf(float v) {
  return 0.5f * v * (1.0f + erff(v * 0.70710678118654752f));
}

// ---------------------------------------------------------------------------
// GEMM: C[M,N](f32) = A[M,K](bf16,row-major) x B[K,N](bf16,row-major) + bias
// act: 0=none, 1=erf-GELU. Optional residual add (f32 [M,N]).
// Optional f32 and/or bf16 outputs.
// ---------------------------------------------------------------------------
__global__ __launch_bounds__(128)
void gemm_bf16_wmma(const bf16_t* __restrict__ A, const bf16_t* __restrict__ Bm,
                    const float* __restrict__ bias, const float* __restrict__ resid,
                    float* __restrict__ outF, bf16_t* __restrict__ outB,
                    int M, int N, int K, int act) {
  __shared__ __align__(32) bf16_t sA[64 * 32];   // 4 A fragments (tiles of rows)
  __shared__ __align__(32) bf16_t sB[64 * 32];   // 4 B fragments (tiles of cols)

  const int tid  = threadIdx.x;
  const int wave = tid >> 5;
  const int lane = tid & 31;
  const int mBase = blockIdx.y * 64;
  const int nBase = blockIdx.x * 64;

  v8f acc[4] = {};

  for (int k0 = 0; k0 < K; k0 += 32) {
    // ---- stage A tile 64x32 into fragment-major LDS ----
    if ((mBase + 64 <= M) && (k0 + 32 <= K) && ((K & 7) == 0)) {
      // vector path: 8 consecutive K elems land in one lane, contiguous e.
      for (int i = tid; i < 256; i += 128) {
        int r = i >> 2, c8 = (i & 3) << 3;
        uint4 v = *(const uint4*)(A + (size_t)(mBase + r) * K + k0 + c8);
        int lane_d = (r & 15) + ((c8 & 8) << 1);
        int e      = (c8 & 16) >> 1;
        *(uint4*)&sA[(r >> 4) * 512 + lane_d * 16 + e] = v;
      }
    } else {
      for (int i = tid; i < 2048; i += 128) {
        int r = i >> 5, kk = i & 31;
        int gr = mBase + r, gc = k0 + kk;
        bf16_t v = (gr < M && gc < K) ? A[(size_t)gr * K + gc] : (bf16_t)0.0f;
        sA[a_off(r, kk)] = v;
      }
    }
    // ---- stage B tile 32x64 into fragment-major LDS ----
    if ((nBase + 64 <= N) && (k0 + 32 <= K) && ((N & 7) == 0)) {
      for (int i = tid; i < 256; i += 128) {
        int kk = i >> 3, n8 = (i & 7) << 3;
        uint4 v = *(const uint4*)(Bm + (size_t)(k0 + kk) * N + nBase + n8);
        const bf16_t* pv = (const bf16_t*)&v;
        int e = frag_e(kk), la = frag_lane_add(kk);
        int tb = (n8 >> 4) * 512, nb = n8 & 15;
#pragma unroll
        for (int j = 0; j < 8; ++j)
          sB[tb + (nb + j + la) * 16 + e] = pv[j];
      }
    } else {
      for (int i = tid; i < 2048; i += 128) {
        int kk = i >> 6, n = i & 63;
        int gk = k0 + kk, gc = nBase + n;
        bf16_t v = (gk < K && gc < N) ? Bm[(size_t)gk * N + gc] : (bf16_t)0.0f;
        sB[b_off(kk, n)] = v;
      }
    }
    __syncthreads();

    const v16bf a = *(const v16bf*)&sA[wave * 512 + lane * 16];
#pragma unroll
    for (int tn = 0; tn < 4; ++tn) {
      const v16bf b = *(const v16bf*)&sB[tn * 512 + lane * 16];
      acc[tn] = __builtin_amdgcn_wmma_f32_16x16x32_bf16(
          false, a, false, b, (short)0, acc[tn], false, false);
    }
    __syncthreads();
  }

  // epilogue: C/D layout — lane col = lane&15 (+16*tn), vgpr i -> row i (+8 if lane>=16)
#pragma unroll
  for (int tn = 0; tn < 4; ++tn) {
    const int col = nBase + (tn << 4) + (lane & 15);
    if (col >= N) continue;
    const float bv = bias ? bias[col] : 0.0f;
#pragma unroll
    for (int vi = 0; vi < 8; ++vi) {
      const int row = mBase + (wave << 4) + vi + ((lane & 16) ? 8 : 0);
      if (row >= M) continue;
      float v = acc[tn][vi] + bv;
      if (act == 1) v = gelu_erf(v);
      if (resid) v += resid[(size_t)row * N + col];
      if (outF) outF[(size_t)row * N + col] = v;
      if (outB) outB[(size_t)row * N + col] = (bf16_t)v;
    }
  }
}

// ---------------------------------------------------------------------------
// Window attention: one block per (window, head). 49 tokens padded to 64,
// head_dim = 32 for every stage. qkv is the fused [B*L, 3C] f32 buffer.
// Q/K^T/V/P all live in LDS in fragment-major layout.
// ---------------------------------------------------------------------------
__global__ __launch_bounds__(128)
void attn_wmma(const float* __restrict__ qkv, const float* __restrict__ rpb,
               bf16_t* __restrict__ outb, int L, int C, int nh, int Wd,
               int nWinW, int nWin) {
  __shared__ __align__(32) bf16_t sQ[64 * 32];   // A-operand frags, 4 row tiles
  __shared__ __align__(32) bf16_t sKT[64 * 32];  // B-operand frags, 4 col tiles
  __shared__ __align__(32) bf16_t sV[64 * 32];   // B-operand frags, 2 ks x 2 col tiles
  __shared__ float  sS[64 * 64];                 // scores, row-major f32
  __shared__ __align__(32) bf16_t sP[64 * 64];   // A-operand frags, 4 strips x 2 ks

  const int tid  = threadIdx.x;
  const int wave = tid >> 5;
  const int lane = tid & 31;

  const int h   = blockIdx.x % nh;
  const int wid = blockIdx.x / nh;
  const int b   = wid / nWin;
  const int wi  = wid % nWin;
  const int wr  = wi / nWinW;
  const int wc  = wi % nWinW;

  const float scale = 0.17677669529663687f;  // 32^-0.5
  const int   C3    = 3 * C;

  // stage Q (pre-scaled), K^T, V into fragment-major LDS; pad tokens 49..63
  for (int i = tid; i < 64 * 32; i += 128) {
    int t = i >> 5, d = i & 31;
    float q = 0.f, k = 0.f, v = 0.f;
    if (t < 49) {
      int r = t / 7, c2 = t % 7;
      int gl = (wr * 7 + r) * Wd + (wc * 7 + c2);
      size_t base = (size_t)(b * L + gl) * C3 + h * 32 + d;
      q = qkv[base] * scale;
      k = qkv[base + C];
      v = qkv[base + 2 * C];
    }
    sQ[a_off(t, d)] = (bf16_t)q;            // A: row=t, k=d
    sKT[b_off(d, t)] = (bf16_t)k;           // B: k=d,  col=t
    // V as B-operand over two K-steps: k=t (0..63), n=d (0..31)
    sV[((t >> 5) << 10) + ((d >> 4) << 9) +
       ((d & 15) + frag_lane_add(t & 31)) * 16 + frag_e(t & 31)] = (bf16_t)v;
  }
  __syncthreads();

  // S = Q K^T  (single K-step of 32), + relative position bias
  {
    const v16bf a = *(const v16bf*)&sQ[wave * 512 + lane * 16];
#pragma unroll
    for (int tn = 0; tn < 4; ++tn) {
      const v16bf bfr = *(const v16bf*)&sKT[tn * 512 + lane * 16];
      v8f c = {};
      c = __builtin_amdgcn_wmma_f32_16x16x32_bf16(false, a, false, bfr,
                                                  (short)0, c, false, false);
      const int col = (tn << 4) + (lane & 15);
#pragma unroll
      for (int vi = 0; vi < 8; ++vi) {
        const int row = (wave << 4) + vi + ((lane & 16) ? 8 : 0);
        float s = c[vi];
        if (row < 49) {
          if (col < 49) {
            int ri = row / 7, ci = row % 7, rj = col / 7, cj = col % 7;
            int idx = (ri - rj + 6) * 13 + (ci - cj + 6);
            s += rpb[(size_t)idx * nh + h];
          } else {
            s = -1e30f;  // padded key columns
          }
        } else {
          s = 0.0f;      // padded query rows: harmless uniform row
        }
        sS[row * 64 + col] = s;
      }
    }
  }
  __syncthreads();

  // softmax per row; P written bf16 directly in A-operand fragment layout
  if (tid < 64) {
    const int row = tid;
    float m = -1e30f;
    for (int j = 0; j < 64; ++j) m = fmaxf(m, sS[row * 64 + j]);
    float sum = 0.f;
    for (int j = 0; j < 64; ++j) sum += expf(sS[row * 64 + j] - m);
    const float inv = 1.0f / sum;
    const int strip = (row >> 4) << 10;  // 2 ks-frags of 512 per row-strip
    for (int j = 0; j < 64; ++j) {
      float p = expf(sS[row * 64 + j] - m) * inv;
      int kk = j & 31;
      sP[strip + ((j >> 5) << 9) +
         ((row & 15) + frag_lane_add(kk)) * 16 + frag_e(kk)] = (bf16_t)p;
    }
  }
  __syncthreads();

  // O = P V  (64x64 x 64x32, two K-steps)
  v8f o[2] = {};
#pragma unroll
  for (int ks = 0; ks < 2; ++ks) {
    const v16bf a = *(const v16bf*)&sP[(wave << 10) + (ks << 9) + lane * 16];
#pragma unroll
    for (int tn = 0; tn < 2; ++tn) {
      const v16bf bfr = *(const v16bf*)&sV[(ks << 10) + (tn << 9) + lane * 16];
      o[tn] = __builtin_amdgcn_wmma_f32_16x16x32_bf16(false, a, false, bfr,
                                                      (short)0, o[tn], false, false);
    }
  }

  // scatter valid tokens back to [B*L, C] bf16
#pragma unroll
  for (int tn = 0; tn < 2; ++tn) {
    const int d = (tn << 4) + (lane & 15);
#pragma unroll
    for (int vi = 0; vi < 8; ++vi) {
      const int t = (wave << 4) + vi + ((lane & 16) ? 8 : 0);
      if (t < 49) {
        int r = t / 7, c2 = t % 7;
        int gl = (wr * 7 + r) * Wd + (wc * 7 + c2);
        outb[(size_t)(b * L + gl) * C + h * 32 + d] = (bf16_t)o[tn][vi];
      }
    }
  }
}

// ---------------------------------------------------------------------------
// LayerNorm over last dim C; emits bf16 (GEMM operand).
// ---------------------------------------------------------------------------
__global__ __launch_bounds__(128)
void ln_bf16(const float* __restrict__ x, const float* __restrict__ g,
             const float* __restrict__ b, bf16_t* __restrict__ out, int C) {
  __shared__ float red[128];
  const int row = blockIdx.x;
  const int tid = threadIdx.x;
  const float* xr = x + (size_t)row * C;

  float s = 0.f;
  for (int c = tid; c < C; c += 128) s += xr[c];
  red[tid] = s; __syncthreads();
  for (int st = 64; st > 0; st >>= 1) {
    if (tid < st) red[tid] += red[tid + st];
    __syncthreads();
  }
  const float mean = red[0] / C; __syncthreads();

  float v = 0.f;
  for (int c = tid; c < C; c += 128) { float d = xr[c] - mean; v += d * d; }
  red[tid] = v; __syncthreads();
  for (int st = 64; st > 0; st >>= 1) {
    if (tid < st) red[tid] += red[tid + st];
    __syncthreads();
  }
  const float inv = rsqrtf(red[0] / C + 1e-5f);

  for (int c = tid; c < C; c += 128)
    out[(size_t)row * C + c] = (bf16_t)((xr[c] - mean) * inv * g[c] + b[c]);
}

// ---------------------------------------------------------------------------
// Training-mode BatchNorm over M=(B*H*W) per channel n.
// ---------------------------------------------------------------------------
__global__ __launch_bounds__(256)
void bn_stats(const float* __restrict__ x, int M, int N,
              float* __restrict__ mean, float* __restrict__ var) {
  __shared__ float rs[256], rq[256];
  const int n = blockIdx.x, tid = threadIdx.x;
  float s = 0.f, q = 0.f;
  for (int m = tid; m < M; m += 256) {
    float v = x[(size_t)m * N + n];
    s += v; q += v * v;
  }
  rs[tid] = s; rq[tid] = q; __syncthreads();
  for (int st = 128; st > 0; st >>= 1) {
    if (tid < st) { rs[tid] += rs[tid + st]; rq[tid] += rq[tid + st]; }
    __syncthreads();
  }
  if (tid == 0) {
    float mu = rs[0] / M;
    mean[n] = mu;
    var[n] = rq[0] / M - mu * mu;
  }
}

__global__ void bn_apply_relu(const float* __restrict__ x,
                              const float* __restrict__ mean,
                              const float* __restrict__ var,
                              const float* __restrict__ g,
                              const float* __restrict__ b,
                              float* __restrict__ outF, bf16_t* __restrict__ outB,
                              int M, int N) {
  const size_t total = (size_t)M * N;
  for (size_t i = blockIdx.x * (size_t)blockDim.x + threadIdx.x; i < total;
       i += (size_t)gridDim.x * blockDim.x) {
    int n = (int)(i % N);
    float v = (x[i] - mean[n]) * rsqrtf(var[n] + 1e-5f) * g[n] + b[n];
    v = fmaxf(v, 0.0f);
    outF[i] = v;
    if (outB) outB[i] = (bf16_t)v;
  }
}

// ---------------------------------------------------------------------------
// f32 -> bf16 copy (weights stored [K,N] row-major already)
// ---------------------------------------------------------------------------
__global__ void cvt_f32_bf16(const float* __restrict__ src, bf16_t* __restrict__ dst,
                             size_t n) {
  for (size_t i = blockIdx.x * (size_t)blockDim.x + threadIdx.x; i < n;
       i += (size_t)gridDim.x * blockDim.x)
    dst[i] = (bf16_t)src[i];
}

// conv weight [OC,IC,KH,KW] f32 -> GEMM B [IC*KH*KW, OC] bf16
__global__ void repack_conv_w(const float* __restrict__ w, bf16_t* __restrict__ dst,
                              int OC, int IC, int KH, int KW) {
  const size_t total = (size_t)OC * IC * KH * KW;
  for (size_t i = blockIdx.x * (size_t)blockDim.x + threadIdx.x; i < total;
       i += (size_t)gridDim.x * blockDim.x) {
    int kw = (int)(i % KW);
    int kh = (int)((i / KW) % KH);
    int ci = (int)((i / (KW * KH)) % IC);
    int oc = (int)(i / ((size_t)KW * KH * IC));
    dst[((size_t)(ci * KH * KW + kh * KW + kw)) * OC + oc] = (bf16_t)w[i];
  }
}

// patch-embed im2col: x NCHW [8,3,224,224] -> A bf16 [8*56*56, 48]
__global__ void im2col_patch(const float* __restrict__ x, bf16_t* __restrict__ dst) {
  const size_t total = (size_t)8 * 56 * 56 * 48;
  for (size_t i = blockIdx.x * (size_t)blockDim.x + threadIdx.x; i < total;
       i += (size_t)gridDim.x * blockDim.x) {
    int k = (int)(i % 48);
    size_t m = i / 48;
    int ci = k / 16, kh = (k % 16) / 4, kw = k % 4;
    int b = (int)(m / 3136);
    int t = (int)(m % 3136);
    int oh = t / 56, ow = t % 56;
    dst[i] = (bf16_t)x[(((size_t)b * 3 + ci) * 224 + oh * 4 + kh) * 224 + ow * 4 + kw];
  }
}

// generic im2col from token layout [B, H*W, C] f32 -> A bf16 [B*OH*OW, C*KH*KW]
__global__ void im2col_tok(const float* __restrict__ src, bf16_t* __restrict__ dst,
                           int Bn, int H, int W, int C, int KH, int KW,
                           int stride, int pad, int OH, int OW) {
  const int K = C * KH * KW;
  const size_t total = (size_t)Bn * OH * OW * K;
  for (size_t i = blockIdx.x * (size_t)blockDim.x + threadIdx.x; i < total;
       i += (size_t)gridDim.x * blockDim.x) {
    int k = (int)(i % K);
    size_t m = i / K;
    int kw = k % KW, kh = (k / KW) % KH, ci = k / (KW * KH);
    int b = (int)(m / ((size_t)OH * OW));
    int t = (int)(m % ((size_t)OH * OW));
    int oh = t / OW, ow = t % OW;
    int ih = oh * stride + kh - pad;
    int iw = ow * stride + kw - pad;
    float v = 0.0f;
    if (ih >= 0 && ih < H && iw >= 0 && iw < W)
      v = src[((size_t)b * H * W + ih * W + iw) * C + ci];
    dst[i] = (bf16_t)v;
  }
}

// ---------------------------------------------------------------------------
// Host orchestration
// ---------------------------------------------------------------------------
extern "C" void kernel_launch(void* const* d_in, const int* in_sizes, int n_in,
                              void* d_out, int out_size, void* d_ws, size_t ws_size,
                              hipStream_t stream) {
  (void)in_sizes; (void)n_in; (void)out_size; (void)ws_size;

  // --- parameter pointers ---------------------------------------------------
  // Assumed flattening: d_in[0] = x (top-level setup_inputs order), then the
  // params pytree in JAX tree-flatten order (dict keys sorted at each level,
  // lists in order):
  //   head{bn1_b,bn1_g,bn2_b,bn2_g,c1_b,c1_w,c2_b,c2_w,cf_b,cf_w},
  //   patch_b, patch_w,
  //   stages[i]{ blocks[d]{fc1_b,fc1_w,fc2_b,fc2_w,n1b,n1g,n2b,n2g,
  //                        proj_b,proj_w,qkv_b,qkv_w,rpb}, down_b, down_w }
  auto F = [&](int i) { return (const float*)d_in[i]; };
  const float* x_in = F(0);
  int pi = 1;
  const float* hbn1b = F(pi++); const float* hbn1g = F(pi++);
  const float* hbn2b = F(pi++); const float* hbn2g = F(pi++);
  const float* hc1b  = F(pi++); const float* hc1w  = F(pi++);
  const float* hc2b  = F(pi++); const float* hc2w  = F(pi++);
  const float* hcfb  = F(pi++); const float* hcfw  = F(pi++);
  const float* patch_b = F(pi++);
  const float* patch_w = F(pi++);

  struct Blk {
    const float *fc1_b, *fc1_w, *fc2_b, *fc2_w, *n1b, *n1g, *n2b, *n2g;
    const float *proj_b, *proj_w, *qkv_b, *qkv_w, *rpb;
  };
  static const int DEPTHS_[4] = {2, 2, 6, 2};
  static const int HEADS_[4]  = {4, 8, 16, 32};
  Blk blks[12];
  const float* down_b[3];
  const float* down_w[3];
  {
    int bi = 0;
    for (int s = 0; s < 4; ++s) {
      for (int d = 0; d < DEPTHS_[s]; ++d) {
        Blk& bp = blks[bi++];
        bp.fc1_b = F(pi++); bp.fc1_w = F(pi++);
        bp.fc2_b = F(pi++); bp.fc2_w = F(pi++);
        bp.n1b = F(pi++); bp.n1g = F(pi++);
        bp.n2b = F(pi++); bp.n2g = F(pi++);
        bp.proj_b = F(pi++); bp.proj_w = F(pi++);
        bp.qkv_b = F(pi++); bp.qkv_w = F(pi++);
        bp.rpb = F(pi++);
      }
      if (s < 3) { down_b[s] = F(pi++); down_w[s] = F(pi++); }
    }
  }

  // --- workspace arena ------------------------------------------------------
  char* wsb = (char*)d_ws;
  size_t off = 0;
  auto alloc = [&](size_t bytes) {
    off = (off + 255) & ~(size_t)255;
    void* p = wsb + off;
    off += bytes;
    return p;
  };
  const size_t MAXTOK = (size_t)8 * 3136 * 128;  // max B*L*C = 3,211,264
  float*  tF    = (float*)alloc(MAXTOK * 4);
  float*  tG    = (float*)alloc(MAXTOK * 4);
  bf16_t* lnb   = (bf16_t*)alloc(MAXTOK * 2);
  float*  qkvF  = (float*)alloc(MAXTOK * 3 * 4);   // [B*L, 3C]
  bf16_t* attnb = (bf16_t*)alloc(MAXTOK * 2);
  bf16_t* hidb  = (bf16_t*)alloc(MAXTOK * 4 * 2);  // [B*L, 4C]
  bf16_t* colb  = (bf16_t*)alloc((size_t)4000000 * 2);
  bf16_t* wbuf  = (bf16_t*)alloc((size_t)4800000 * 2);
  float*  z1    = (float*)alloc((size_t)392 * 512 * 4);
  float*  z2    = (float*)alloc((size_t)392 * 256 * 4);
  bf16_t* z2b   = (bf16_t*)alloc((size_t)392 * 256 * 2);
  float*  bnm   = (float*)alloc(512 * 4);
  float*  bnv   = (float*)alloc(512 * 4);

  // --- launch helpers -------------------------------------------------------
  auto gemm = [&](const bf16_t* A, const bf16_t* Bw, const float* bias,
                  const float* resid, float* oF, bf16_t* oB,
                  int M, int N, int K, int act) {
    dim3 g((N + 63) / 64, (M + 63) / 64);
    gemm_bf16_wmma<<<g, 128, 0, stream>>>(A, Bw, bias, resid, oF, oB, M, N, K, act);
  };
  auto cvt = [&](const float* s, bf16_t* d, size_t n) {
    int blocks = (int)((n + 255) / 256);
    cvt_f32_bf16<<<blocks, 256, 0, stream>>>(s, d, n);
  };
  auto repack = [&](const float* w, int OC, int IC, int KH, int KW) {
    size_t n = (size_t)OC * IC * KH * KW;
    repack_conv_w<<<(int)((n + 255) / 256), 256, 0, stream>>>(w, wbuf, OC, IC, KH, KW);
  };

  // --- patch embed ----------------------------------------------------------
  {
    size_t n = (size_t)8 * 3136 * 48;
    im2col_patch<<<(int)((n + 255) / 256), 256, 0, stream>>>(x_in, colb);
    repack(patch_w, 128, 3, 4, 4);
    gemm(colb, wbuf, patch_b, nullptr, tF, nullptr, 8 * 3136, 128, 48, 0);
  }

  // --- stages ---------------------------------------------------------------
  int Hc = 56, Wc = 56, C = 128;
  float* cur = tF;
  float* alt = tG;
  int bi = 0;
  for (int s = 0; s < 4; ++s) {
    const int L = Hc * Wc;
    const int M = 8 * L;
    const int nh = HEADS_[s];
    const int nWinW = Wc / 7;
    const int nWin = nWinW * nWinW;
    for (int d = 0; d < DEPTHS_[s]; ++d) {
      const Blk& bp = blks[bi++];
      // LN1 -> bf16
      ln_bf16<<<M, 128, 0, stream>>>(cur, bp.n1g, bp.n1b, lnb, C);
      // qkv GEMM -> f32
      cvt(bp.qkv_w, wbuf, (size_t)C * 3 * C);
      gemm(lnb, wbuf, bp.qkv_b, nullptr, qkvF, nullptr, M, 3 * C, C, 0);
      // window attention -> bf16
      attn_wmma<<<8 * nWin * nh, 128, 0, stream>>>(qkvF, bp.rpb, attnb,
                                                   L, C, nh, Wc, nWinW, nWin);
      // proj GEMM + residual -> alt (f32)
      cvt(bp.proj_w, wbuf, (size_t)C * C);
      gemm(attnb, wbuf, bp.proj_b, cur, alt, nullptr, M, C, C, 0);
      // LN2 -> bf16
      ln_bf16<<<M, 128, 0, stream>>>(alt, bp.n2g, bp.n2b, lnb, C);
      // fc1 GEMM + GELU -> bf16 hidden
      cvt(bp.fc1_w, wbuf, (size_t)C * 4 * C);
      gemm(lnb, wbuf, bp.fc1_b, nullptr, nullptr, hidb, M, 4 * C, C, 1);
      // fc2 GEMM + residual -> cur (f32)
      cvt(bp.fc2_w, wbuf, (size_t)4 * C * C);
      gemm(hidb, wbuf, bp.fc2_b, alt, cur, nullptr, M, C, 4 * C, 0);
    }
    if (s < 3) {
      const int OH = Hc / 2, OW = Wc / 2, C2 = 2 * C;
      size_t n = (size_t)8 * OH * OW * C * 4;
      im2col_tok<<<(int)((n + 255) / 256), 256, 0, stream>>>(
          cur, colb, 8, Hc, Wc, C, 2, 2, 2, 0, OH, OW);
      repack(down_w[s], C2, C, 2, 2);
      gemm(colb, wbuf, down_b[s], nullptr, alt, nullptr, 8 * OH * OW, C2, 4 * C, 0);
      float* t = cur; cur = alt; alt = t;
      Hc = OH; Wc = OW; C = C2;
    }
  }

  // --- detection head (7x7, channel-last layout == token layout) ------------
  {
    const int M = 8 * 49;
    // conv1 3x3 p1: K = 1024*9
    size_t n = (size_t)M * 1024 * 9;
    im2col_tok<<<(int)((n + 255) / 256), 256, 0, stream>>>(
        cur, colb, 8, 7, 7, 1024, 3, 3, 1, 1, 7, 7);
    repack(hc1w, 512, 1024, 3, 3);
    gemm(colb, wbuf, hc1b, nullptr, z1, nullptr, M, 512, 1024 * 9, 0);
    bn_stats<<<512, 256, 0, stream>>>(z1, M, 512, bnm, bnv);
    bn_apply_relu<<<(int)(((size_t)M * 512 + 255) / 256), 256, 0, stream>>>(
        z1, bnm, bnv, hbn1g, hbn1b, z1, nullptr, M, 512);
    // conv2 3x3 p1: K = 512*9
    n = (size_t)M * 512 * 9;
    im2col_tok<<<(int)((n + 255) / 256), 256, 0, stream>>>(
        z1, colb, 8, 7, 7, 512, 3, 3, 1, 1, 7, 7);
    repack(hc2w, 256, 512, 3, 3);
    gemm(colb, wbuf, hc2b, nullptr, z2, nullptr, M, 256, 512 * 9, 0);
    bn_stats<<<256, 256, 0, stream>>>(z2, M, 256, bnm, bnv);
    bn_apply_relu<<<(int)(((size_t)M * 256 + 255) / 256), 256, 0, stream>>>(
        z2, bnm, bnv, hbn2g, hbn2b, z2, z2b, M, 256);
    // conv3 1x1 -> d_out [B,7,7,30] == GEMM [M, 30]
    repack(hcfw, 30, 256, 1, 1);
    gemm(z2b, wbuf, hcfb, nullptr, (float*)d_out, nullptr, M, 30, 256, 0);
  }
}
